// KANLayer_1984274891371
// MI455X (gfx1250) — matrix operand order repaired and checked by reading, not compiled, not run on hardware
//
#include <hip/hip_runtime.h>

// ---------------------------------------------------------------------------
// KAN layer on MI455X (gfx1250):
//   out[b,o] = x @ Wbase^T + einsum('bil,oil->bo', B-spline basis, Wspl)
// Folded into a single f16 WMMA GEMM over augmented K = IN_F * 12:
//   channel 0  : x[b,i]
//   channel 1+l: basis_l(x[b,i]), l = 0..10  (grid_size+order = 11)
// GEMM main loop uses a 3-stage GLOBAL_LOAD_ASYNC_TO_LDS pipeline (ASYNCcnt)
// when the toolchain exposes the gfx1250 async-LDS builtins.
// ---------------------------------------------------------------------------

#define BATCH   4096
#define IN_F    1024
#define OUT_F   1024
#define NCH     12                  // 1 base channel + 11 spline channels
#define KAUG    (IN_F * NCH)        // 12288
#define TK      32                  // WMMA K step (f16) = LDS stage depth
#define LDT     40                  // padded LDS row length in f16 (80 B rows)
#define NSTAGE  (KAUG / TK)         // 384

typedef __attribute__((ext_vector_type(16))) _Float16 v16h;
typedef __attribute__((ext_vector_type(8)))  _Float16 v8h;
typedef __attribute__((ext_vector_type(8)))  float    v8f;
typedef int vi4 __attribute__((vector_size(16)));   // int4, matches builtin param

#if __has_builtin(__builtin_amdgcn_global_load_async_to_lds_b128)
#define HAVE_ASYNC_LDS 1
#else
#define HAVE_ASYNC_LDS 0
#endif

// Builtin signature (from hipcc diagnostic): param0 = int4 addrspace(1)*,
// param1 = int4 addrspace(3)*, then imm offset, imm cpol.
#define GAS(p) ((__attribute__((address_space(1))) vi4*)(p))
#define LAS(p) ((__attribute__((address_space(3))) vi4*)(p))

// Uniform extended knot vector: grid[j] = (j-3)*h - 1, h = 2/8 = 0.25, j=0..14
__device__ __forceinline__ float knot(int j) {
    return (float)(j - 3) * 0.25f - 1.0f;
}

// ---------------------------------------------------------------------------
// Kernel A: augmented features  feat[b][i*12 + c]  (f16)
// Cox-de Boor recursion, degree 3, uniform knots (denominators = d*h > 0).
// ---------------------------------------------------------------------------
__global__ __launch_bounds__(256)
void kan_features(const float* __restrict__ x, _Float16* __restrict__ feat) {
    const int idx = blockIdx.x * 256 + threadIdx.x;   // idx = b*IN_F + i
    if (idx >= BATCH * IN_F) return;
    const float xv = x[idx];

    float b0[14];
#pragma unroll
    for (int j = 0; j < 14; ++j)
        b0[j] = (xv >= knot(j) && xv < knot(j + 1)) ? 1.0f : 0.0f;

    float b1[13];
#pragma unroll
    for (int j = 0; j < 13; ++j)   // 1/(1*h) = 4
        b1[j] = (xv - knot(j)) * 4.0f * b0[j] + (knot(j + 2) - xv) * 4.0f * b0[j + 1];

    float b2[12];
#pragma unroll
    for (int j = 0; j < 12; ++j)   // 1/(2*h) = 2
        b2[j] = (xv - knot(j)) * 2.0f * b1[j] + (knot(j + 3) - xv) * 2.0f * b1[j + 1];

    float b3[11];
#pragma unroll
    for (int j = 0; j < 11; ++j)   // 1/(3*h) = 4/3
        b3[j] = (xv - knot(j)) * (4.0f / 3.0f) * b2[j] +
                (knot(j + 4) - xv) * (4.0f / 3.0f) * b2[j + 1];

    union { _Float16 h[NCH]; uint2 u2[3]; } pk;
    pk.h[0] = (_Float16)xv;
#pragma unroll
    for (int j = 0; j < 11; ++j) pk.h[1 + j] = (_Float16)b3[j];

    uint2* dst = (uint2*)(feat + (size_t)idx * NCH);  // 24-byte stride, 8B aligned
    dst[0] = pk.u2[0]; dst[1] = pk.u2[1]; dst[2] = pk.u2[2];
}

// ---------------------------------------------------------------------------
// Kernel B: pack f16 weights  w16[o][i*12 + c]
// ---------------------------------------------------------------------------
__global__ __launch_bounds__(256)
void kan_pack_w(const float* __restrict__ base_w, const float* __restrict__ spl_w,
                _Float16* __restrict__ w16) {
    const int idx = blockIdx.x * 256 + threadIdx.x;   // idx = o*IN_F + i
    if (idx >= OUT_F * IN_F) return;

    union { _Float16 h[NCH]; uint2 u2[3]; } pk;
    pk.h[0] = (_Float16)base_w[idx];
    const float* sp = spl_w + (size_t)idx * 11;
#pragma unroll
    for (int l = 0; l < 11; ++l) pk.h[1 + l] = (_Float16)sp[l];

    uint2* dst = (uint2*)(w16 + (size_t)idx * NCH);
    dst[0] = pk.u2[0]; dst[1] = pk.u2[1]; dst[2] = pk.u2[2];
}

// ---------------------------------------------------------------------------
// Fragment assembly in this code's own contiguous k order (valid: A and B use the same order)
// A 16x32 f16: lane holds row M=lane%16, K = (lane/16)*16 + 0..15 contiguous
// B 32x16 f16: lane holds column N=lane%16, K = (lane/16)*16 + 0..15 contiguous
// ---------------------------------------------------------------------------
__device__ __forceinline__ v16h load_a_frag(const _Float16* rp, int sel) {
    v8h lo = *(const v8h*)(rp + sel * 16);
    v8h hi = *(const v8h*)(rp + sel * 16 + 8);
    return __builtin_shufflevector(lo, hi, 0, 1, 2, 3, 4, 5, 6, 7,
                                   8, 9, 10, 11, 12, 13, 14, 15);
}
__device__ __forceinline__ v16h load_b_frag(const _Float16* rp) {
    v8h lo = *(const v8h*)(rp);
    v8h hi = *(const v8h*)(rp + 8);
    return __builtin_shufflevector(lo, hi, 0, 1, 2, 3, 4, 5, 6, 7,
                                   8, 9, 10, 11, 12, 13, 14, 15);
}

#if HAVE_ASYNC_LDS
__device__ __forceinline__ void wait_async_le4() {
#if __has_builtin(__builtin_amdgcn_s_wait_asynccnt)
    __builtin_amdgcn_s_wait_asynccnt(4);
#else
    asm volatile("s_wait_asynccnt 0x4" ::: "memory");
#endif
}
__device__ __forceinline__ void wait_async_le0() {
#if __has_builtin(__builtin_amdgcn_s_wait_asynccnt)
    __builtin_amdgcn_s_wait_asynccnt(0);
#else
    asm volatile("s_wait_asynccnt 0x0" ::: "memory");
#endif
}
#endif

// ---------------------------------------------------------------------------
// Kernel C: out(4096x1024 f32) = feat(4096xK f16) * w16(1024xK f16)^T
// Workgroup tile 128x128, 8 waves (wave32), each wave 64x32 = 4x2 WMMA tiles.
// ---------------------------------------------------------------------------
__global__ __launch_bounds__(256)
void kan_gemm(const _Float16* __restrict__ feat, const _Float16* __restrict__ w16,
              float* __restrict__ out) {
    const int tid    = threadIdx.x;
    const int lane   = tid & 31;
    const int wave   = tid >> 5;
    const int waveM  = wave >> 2;          // 0..1  (64-row slab)
    const int waveN  = wave & 3;           // 0..3  (32-col slab)
    const int laneLo = lane & 15;
    const int sel    = lane >> 4;          // 0 or 1

    const int m0 = blockIdx.y * 128;
    const int n0 = blockIdx.x * 128;

    // cooperative load mapping: 256 threads cover 128 rows x 32 f16 (64 B/row)
    const int lrow = tid >> 1;             // 0..127
    const int lcol = (tid & 1) * 16;       // 0 or 16 (f16 units)
    const int lofs = lrow * LDT + lcol;

    const _Float16* aG = feat + (size_t)(m0 + lrow) * KAUG + lcol;
    const _Float16* bG = w16  + (size_t)(n0 + lrow) * KAUG + lcol;

    v8f acc[4][2];
#pragma unroll
    for (int mt = 0; mt < 4; ++mt)
#pragma unroll
        for (int nt = 0; nt < 2; ++nt) acc[mt][nt] = (v8f)(0.0f);

#if HAVE_ASYNC_LDS
    // ---- 3-stage async pipeline: LDS written directly by ASYNC loads -------
    __shared__ __align__(16) _Float16 As[3][128 * LDT];   // 3 x 10 KiB
    __shared__ __align__(16) _Float16 Bs[3][128 * LDT];   // 3 x 10 KiB (total 61.4 KiB)

    auto issue = [&](int s, int buf) {
        const _Float16* ag = aG + (size_t)s * TK;
        const _Float16* bg = bG + (size_t)s * TK;
        _Float16* al = &As[buf][lofs];
        _Float16* bl = &Bs[buf][lofs];
        // 4 async b128 per thread per stage (in-order completion per wave)
        __builtin_amdgcn_global_load_async_to_lds_b128(GAS(ag),     LAS(al),     0, 0);
        __builtin_amdgcn_global_load_async_to_lds_b128(GAS(ag + 8), LAS(al + 8), 0, 0);
        __builtin_amdgcn_global_load_async_to_lds_b128(GAS(bg),     LAS(bl),     0, 0);
        __builtin_amdgcn_global_load_async_to_lds_b128(GAS(bg + 8), LAS(bl + 8), 0, 0);
    };

    issue(0, 0);
    issue(1, 1);

    for (int s = 0; s < NSTAGE; ++s) {
        const int buf = s % 3;
        // Current stage's 4 asyncs complete; next stage's 4 may stay in flight.
        if (s + 1 < NSTAGE) wait_async_le4();
        else                wait_async_le0();
        __syncthreads();                    // cross-wave LDS visibility; also
                                            // guarantees buf(s+2) free for reuse
        if (s + 2 < NSTAGE) issue(s + 2, (s + 2) % 3);

        const _Float16* Ab = As[buf];
        const _Float16* Bb = Bs[buf];
        v16h af[4], bf[2];
#pragma unroll
        for (int mt = 0; mt < 4; ++mt)
            af[mt] = load_a_frag(Ab + (waveM * 64 + mt * 16 + laneLo) * LDT, sel);
#pragma unroll
        for (int nt = 0; nt < 2; ++nt)
            bf[nt] = load_b_frag(Bb + (waveN * 32 + nt * 16 + laneLo) * LDT + sel * 16);

#pragma unroll
        for (int mt = 0; mt < 4; ++mt)
#pragma unroll
            for (int nt = 0; nt < 2; ++nt)
                acc[mt][nt] = __builtin_amdgcn_wmma_f32_16x16x32_f16(
                    false, af[mt], false, bf[nt], (short)0, acc[mt][nt],
                    false, false);
    }
#else
    // ---- fallback: synchronous single-buffer staging (round-1 path) --------
    __shared__ __align__(16) _Float16 As[128 * LDT];
    __shared__ __align__(16) _Float16 Bs[128 * LDT];
    _Float16* asw = &As[lofs];
    _Float16* bsw = &Bs[lofs];
    const _Float16* aPtr = aG;
    const _Float16* bPtr = bG;

    for (int k0 = 0; k0 < KAUG; k0 += TK) {
        const uint4 av0 = ((const uint4*)aPtr)[0];
        const uint4 av1 = ((const uint4*)aPtr)[1];
        const uint4 bv0 = ((const uint4*)bPtr)[0];
        const uint4 bv1 = ((const uint4*)bPtr)[1];
        aPtr += TK; bPtr += TK;

        __syncthreads();
        ((uint4*)asw)[0] = av0; ((uint4*)asw)[1] = av1;
        ((uint4*)bsw)[0] = bv0; ((uint4*)bsw)[1] = bv1;
        __syncthreads();

        v16h af[4], bf[2];
#pragma unroll
        for (int mt = 0; mt < 4; ++mt)
            af[mt] = load_a_frag(&As[(waveM * 64 + mt * 16 + laneLo) * LDT], sel);
#pragma unroll
        for (int nt = 0; nt < 2; ++nt)
            bf[nt] = load_b_frag(&Bs[(waveN * 32 + nt * 16 + laneLo) * LDT + sel * 16]);

#pragma unroll
        for (int mt = 0; mt < 4; ++mt)
#pragma unroll
            for (int nt = 0; nt < 2; ++nt)
                acc[mt][nt] = __builtin_amdgcn_wmma_f32_16x16x32_f16(
                    false, af[mt], false, bf[nt], (short)0, acc[mt][nt],
                    false, false);
    }
#endif

    // Epilogue. C/D layout: VGPR r, lanes 0-15 -> (M=r, N=lane);
    // lanes 16-31 -> (M=8+r, N=lane-16).
#pragma unroll
    for (int mt = 0; mt < 4; ++mt) {
#pragma unroll
        for (int nt = 0; nt < 2; ++nt) {
            const int row = m0 + waveM * 64 + mt * 16 + sel * 8;
            const int col = n0 + waveN * 32 + nt * 16 + laneLo;
            float* op = out + (size_t)row * OUT_F + col;
#pragma unroll
            for (int r = 0; r < 8; ++r)
                op[(size_t)r * OUT_F] = acc[mt][nt][r];
        }
    }
}

// ---------------------------------------------------------------------------
// Harness entry point.
// d_in order: x (4096*1024 f32), base_weight (1024*1024 f32),
//             spline_weight (1024*1024*11 f32), grid (15 f32, uniform -> hardcoded)
// d_out: 4096*1024 f32.
// ---------------------------------------------------------------------------
extern "C" void kernel_launch(void* const* d_in, const int* in_sizes, int n_in,
                              void* d_out, int out_size, void* d_ws, size_t ws_size,
                              hipStream_t stream) {
    (void)in_sizes; (void)n_in; (void)out_size; (void)ws_size;

    const float* x      = (const float*)d_in[0];
    const float* base_w = (const float*)d_in[1];
    const float* spl_w  = (const float*)d_in[2];
    // d_in[3] = grid: uniform with h = 0.25, reproduced in-kernel.

    _Float16* feat = (_Float16*)d_ws;                                 // 96 MiB
    _Float16* w16  = (_Float16*)((char*)d_ws + (size_t)BATCH * KAUG * sizeof(_Float16));
    float*    out  = (float*)d_out;

    kan_features<<<(BATCH * IN_F + 255) / 256, 256, 0, stream>>>(x, feat);
    kan_pack_w  <<<(OUT_F * IN_F + 255) / 256, 256, 0, stream>>>(base_w, spl_w, w16);
    kan_gemm    <<<dim3(OUT_F / 128, BATCH / 128), 256, 0, stream>>>(feat, w16, out);
}
